// F_MoE_Block_38946763440636
// MI455X (gfx1250) — compile-verified, matrix operand order checked by reading, not run on hardware
//
#include <hip/hip_runtime.h>
#include <math.h>

typedef __attribute__((ext_vector_type(16))) _Float16 v16h;
typedef __attribute__((ext_vector_type(8)))  _Float16 v8h;
typedef __attribute__((ext_vector_type(4)))  _Float16 v4h;
typedef __attribute__((ext_vector_type(2)))  _Float16 v2h;
typedef __attribute__((ext_vector_type(8)))  float    v8f;

#define B_DIM 8
#define C_DIM 128
#define H_DIM 128
#define W_DIM 128
#define HW_DIM 16384
#define E_DIM 4

#define WSTR 136   // W LDS row stride in halves (= 272 B: 16B-aligned rows, banks spread)
#define KPSTR 68   // XT row stride in v2h units (= 272 B: 16B-aligned, conflict-free reads)
#define NT   64    // N tile per workgroup

// ---------------------------------------------------------------------------
__global__ void zero_kernel(float* __restrict__ p, int n) {
    for (int i = threadIdx.x + blockIdx.x * blockDim.x; i < n; i += blockDim.x * gridDim.x)
        p[i] = 0.0f;
}

// ---------------------------------------------------------------------------
// f32 -> f16 conversion (fp32 expert accumulator -> WMMA operand buffer)
// ---------------------------------------------------------------------------
__global__ void f32_to_f16_kernel(const float* __restrict__ src,
                                  _Float16* __restrict__ dst) {
    const size_t i = ((size_t)blockIdx.x * blockDim.x + threadIdx.x) * 4;
    float4 v = *(const float4*)(src + i);
    v4h h;
    h[0] = (_Float16)v.x; h[1] = (_Float16)v.y;
    h[2] = (_Float16)v.z; h[3] = (_Float16)v.w;
    *(v4h*)(dst + i) = h;
}

// ---------------------------------------------------------------------------
// LayerNorm over channels (per pixel) -> f16 x_in, plus per-(b,c) partial sums
// for the router's global average pool.
// ---------------------------------------------------------------------------
__global__ void ln_kernel(const float* __restrict__ x,
                          const float* __restrict__ gamma,
                          const float* __restrict__ beta,
                          _Float16* __restrict__ xin,
                          float* __restrict__ feat_sum) {
    __shared__ float sh[C_DIM];
    const int tid = threadIdx.x;                 // 128
    const int b   = blockIdx.z;
    const int pix = blockIdx.x * 128 + tid;
    const float* xb = x + (size_t)b * C_DIM * HW_DIM + pix;

    if (tid < C_DIM) sh[tid] = 0.0f;
    __syncthreads();

    float s = 0.0f, ss = 0.0f;
    #pragma unroll 8
    for (int c = 0; c < C_DIM; ++c) {
        float v = xb[(size_t)c * HW_DIM];
        s += v; ss += v * v;
    }
    const float mu  = s * (1.0f / C_DIM);
    const float var = ss * (1.0f / C_DIM) - mu * mu;
    const float inv = 1.0f / sqrtf(var + 1e-6f);

    _Float16* xo = xin + (size_t)b * C_DIM * HW_DIM + pix;
    const int lane = tid & 31;
    for (int c = 0; c < C_DIM; ++c) {
        float v = (xb[(size_t)c * HW_DIM] - mu) * inv * gamma[c] + beta[c];
        xo[(size_t)c * HW_DIM] = (_Float16)v;
        float r = v;
        #pragma unroll
        for (int off = 16; off > 0; off >>= 1)
            r += __shfl_xor(r, off, 32);         // wave32 reduction
        if (lane == 0) atomicAdd(&sh[c], r);
    }
    __syncthreads();
    if (tid < C_DIM) atomicAdd(&feat_sum[b * C_DIM + tid], sh[tid]);
}

// ---------------------------------------------------------------------------
// Router: logits = [feat_avg, prompt] @ rw + rb ; softmax ; top-2 -> gate[B,E]
// ---------------------------------------------------------------------------
__global__ void router_kernel(const float* __restrict__ feat_sum,
                              const float* __restrict__ prompt,
                              const float* __restrict__ rw,
                              const float* __restrict__ rb,
                              float* __restrict__ gate) {
    const int b = threadIdx.x;
    if (b >= B_DIM) return;
    float logits[E_DIM];
    for (int e = 0; e < E_DIM; ++e) {
        float acc = rb[e];
        for (int i = 0; i < C_DIM; ++i)
            acc += (feat_sum[b * C_DIM + i] * (1.0f / HW_DIM)) * rw[i * E_DIM + e];
        for (int i = 0; i < C_DIM; ++i)
            acc += prompt[b * C_DIM + i] * rw[(C_DIM + i) * E_DIM + e];
        logits[e] = acc;
    }
    float m = logits[0];
    for (int e = 1; e < E_DIM; ++e) m = fmaxf(m, logits[e]);
    float p[E_DIM], sum = 0.0f;
    for (int e = 0; e < E_DIM; ++e) { p[e] = expf(logits[e] - m); sum += p[e]; }
    for (int e = 0; e < E_DIM; ++e) p[e] /= sum;
    int i1 = 0;
    for (int e = 1; e < E_DIM; ++e) if (p[e] > p[i1]) i1 = e;
    int i2 = (i1 == 0) ? 1 : 0;
    for (int e = 0; e < E_DIM; ++e) if (e != i1 && p[e] > p[i2]) i2 = e;
    for (int e = 0; e < E_DIM; ++e)
        gate[b * E_DIM + e] = (e == i1 || e == i2) ? p[e] : 0.0f;
}

// ---------------------------------------------------------------------------
// 128x128 GEMM over HW columns via v_wmma_f32_16x16x32_f16.
//   Y[b,o,n] (+)= scale * ( sum_c W[o,c] * X[b,c,n] + bias[o] ) (+ resid[b,o,n])
// 256 threads = 8 wave32s; 128x64 tile; K=128 resident in LDS.
// A fragment  = 2x ds_load_b128 from row-major W tile.
// B fragment  = 2x ds_load_b128 from K-pair-packed transposed X tile:
//               XT[n][kp] = { X[2kp][n], X[2kp+1][n] }  (one dword per entry)
// ---------------------------------------------------------------------------
__global__ __launch_bounds__(256)
void gemm128_wmma(const float* __restrict__ Wg,      // [128,128] row-major (o,c)
                  const _Float16* __restrict__ Xg,   // [B,128,HW]
                  float* __restrict__ Yg,            // [B,128,HW]
                  const float* __restrict__ bias,    // [128]
                  const float* __restrict__ gate,    // nullable; scale = gate[b*gstride]
                  int gstride,
                  const float* __restrict__ resid,   // nullable [B,128,HW]
                  int accumulate) {
    __shared__ __align__(16) _Float16 Wl[C_DIM * WSTR];   // 34.0 KB
    __shared__ __align__(16) v2h      XT[NT * KPSTR];     // 17.0 KB

    const int tid = threadIdx.x;
    const int b   = blockIdx.z;
    const int n0  = blockIdx.x * NT;
    const _Float16* Xb = Xg + (size_t)b * C_DIM * HW_DIM;

    // Stage weights: 128*128 floats -> f16, float4 granularity.
    for (int q = tid; q < C_DIM * 32; q += 256) {
        int r = q >> 5, cq = q & 31;
        float4 wv = *(const float4*)(Wg + r * C_DIM + cq * 4);
        v4h hv;
        hv[0] = (_Float16)wv.x; hv[1] = (_Float16)wv.y;
        hv[2] = (_Float16)wv.z; hv[3] = (_Float16)wv.w;
        *(v4h*)&Wl[r * WSTR + cq * 4] = hv;
    }
    // Stage activations: pair-interleave K rows (2kp, 2kp+1) and transpose
    // into XT[n][kp]. Global reads are 16B along n; interleave done in regs.
    for (int q = tid; q < 64 * 8; q += 256) {     // kp (0..63) x 8-col groups
        const int kp = q >> 3, n = (q & 7) * 8;
        const _Float16* s0 = Xb + (size_t)(2 * kp) * HW_DIM + n0 + n;
        v8h e0 = *(const v8h*)s0;
        v8h e1 = *(const v8h*)(s0 + HW_DIM);
        #pragma unroll
        for (int i = 0; i < 8; ++i) {
            v2h p; p[0] = e0[i]; p[1] = e1[i];
            XT[(n + i) * KPSTR + kp] = p;
        }
    }
    __syncthreads();

    const int wid  = tid >> 5;         // 0..7 -> M strip
    const int lane = tid & 31;
    const int half = (lane >> 4) & 1;
    const int c16  = lane & 15;
    const int m0   = wid * 16;

    v8f acc[4] = {};

    #pragma unroll
    for (int k0 = 0; k0 < C_DIM; k0 += 32) {
        // A: two contiguous 8-half runs (K = k0+8h..+7 and k0+16+8h..+7)
        const _Float16* wr = &Wl[(m0 + c16) * WSTR + k0 + 8 * half];
        v8h alo = *(const v8h*)wr;
        v8h ahi = *(const v8h*)(wr + 16);
        v16h a = __builtin_shufflevector(alo, ahi,
                 0,1,2,3,4,5,6,7,8,9,10,11,12,13,14,15);
        #pragma unroll
        for (int j = 0; j < 4; ++j) {
            const int n = j * 16 + c16;
            // B: two contiguous 4-dword runs of packed K-pairs
            const v2h* br = &XT[n * KPSTR + (k0 >> 1) + 4 * half];
            v8h blo = *(const v8h*)br;
            v8h bhi = *(const v8h*)(br + 8);
            v16h bt = __builtin_shufflevector(blo, bhi,
                      0,1,2,3,4,5,6,7,8,9,10,11,12,13,14,15);
            acc[j] = __builtin_amdgcn_wmma_f32_16x16x32_f16(
                false, a, false, bt, (short)0, acc[j], false, false);
        }
    }

    float scale = 1.0f;
    if (gate) scale = gate[b * gstride];
    float* Yb = Yg + (size_t)b * C_DIM * HW_DIM;
    const float* Rb = resid ? resid + (size_t)b * C_DIM * HW_DIM : nullptr;

    #pragma unroll
    for (int j = 0; j < 4; ++j) {
        const int n = n0 + j * 16 + c16;
        #pragma unroll
        for (int v = 0; v < 8; ++v) {
            const int o = m0 + v + half * 8;          // C/D layout: M = v + 8*(lane>=16)
            float val = (acc[j][v] + bias[o]) * scale;
            if (Rb) val += Rb[(size_t)o * HW_DIM + n];
            const size_t idx = (size_t)o * HW_DIM + n;
            if (accumulate) Yb[idx] += val;
            else            Yb[idx]  = val;
        }
    }
}

// ---------------------------------------------------------------------------
// Depthwise 3x3 SAME conv + bias + exact GELU, f32 in -> f16 out
// ---------------------------------------------------------------------------
__global__ void dwconv_gelu_kernel(const float* __restrict__ y1,
                                   const float* __restrict__ dwk,  // + e*C*9
                                   const float* __restrict__ bdw,  // + e*C
                                   _Float16* __restrict__ g) {
    const int c = blockIdx.y, b = blockIdx.z;
    const int pix = blockIdx.x * blockDim.x + threadIdx.x;
    const int h = pix >> 7, w = pix & 127;
    const float* src = y1 + ((size_t)b * C_DIM + c) * HW_DIM;
    const float* k = dwk + c * 9;
    float acc = bdw[c];
    #pragma unroll
    for (int kh = 0; kh < 3; ++kh) {
        const int hh = h + kh - 1;
        if (hh < 0 || hh >= H_DIM) continue;
        #pragma unroll
        for (int kw = 0; kw < 3; ++kw) {
            const int ww = w + kw - 1;
            if (ww < 0 || ww >= W_DIM) continue;
            acc += src[hh * W_DIM + ww] * k[kh * 3 + kw];
        }
    }
    const float ge = 0.5f * acc * (1.0f + erff(acc * 0.70710678118654752f));
    g[((size_t)b * C_DIM + c) * HW_DIM + pix] = (_Float16)ge;
}

// ---------------------------------------------------------------------------
extern "C" void kernel_launch(void* const* d_in, const int* in_sizes, int n_in,
                              void* d_out, int out_size, void* d_ws, size_t ws_size,
                              hipStream_t stream) {
    const float* x      = (const float*)d_in[0];
    const float* prompt = (const float*)d_in[1];
    const float* gamma  = (const float*)d_in[2];
    const float* beta   = (const float*)d_in[3];
    const float* rw     = (const float*)d_in[4];
    const float* rb     = (const float*)d_in[5];
    const float* w1     = (const float*)d_in[6];
    const float* b1     = (const float*)d_in[7];
    const float* dw     = (const float*)d_in[8];
    const float* bdw    = (const float*)d_in[9];
    const float* w2     = (const float*)d_in[10];
    const float* b2     = (const float*)d_in[11];
    const float* pw     = (const float*)d_in[12];
    const float* pb     = (const float*)d_in[13];
    float* out = (float*)d_out;

    const size_t Nel = (size_t)B_DIM * C_DIM * HW_DIM;
    float*    y1   = (float*)d_ws;                 // [B,C,HW] f32  (64 MB)
    float*    oacc = y1 + Nel;                     // [B,C,HW] f32  (64 MB)
    _Float16* xin  = (_Float16*)(oacc + Nel);      // [B,C,HW] f16  (32 MB)
    _Float16* g    = xin + Nel;                    // [B,C,HW] f16  (32 MB)
    float*    feat = (float*)(g + Nel);            // [B,C]
    float*    gate = feat + B_DIM * C_DIM;         // [B,E]

    zero_kernel<<<4, 256, 0, stream>>>(feat, B_DIM * C_DIM);
    ln_kernel<<<dim3(HW_DIM / 128, 1, B_DIM), 128, 0, stream>>>(x, gamma, beta, xin, feat);
    router_kernel<<<1, 32, 0, stream>>>(feat, prompt, rw, rb, gate);

    const dim3 ggrid(HW_DIM / NT, 1, B_DIM);
    for (int e = 0; e < E_DIM; ++e) {
        gemm128_wmma<<<ggrid, 256, 0, stream>>>(
            w1 + (size_t)e * C_DIM * C_DIM, xin, y1, b1 + e * C_DIM,
            nullptr, 0, nullptr, 0);
        dwconv_gelu_kernel<<<dim3(HW_DIM / 256, C_DIM, B_DIM), 256, 0, stream>>>(
            y1, dw + (size_t)e * C_DIM * 9, bdw + e * C_DIM, g);
        gemm128_wmma<<<ggrid, 256, 0, stream>>>(
            w2 + (size_t)e * C_DIM * C_DIM, g, oacc, b2 + e * C_DIM,
            gate + e, E_DIM, nullptr, e == 0 ? 0 : 1);
    }
    // convert fp32 expert accumulator -> f16 (reuses g, free after the loop)
    f32_to_f16_kernel<<<(unsigned)(Nel / (256 * 4)), 256, 0, stream>>>(oacc, g);
    // final projection + bias + residual -> d_out
    gemm128_wmma<<<ggrid, 256, 0, stream>>>(
        pw, g, out, pb, nullptr, 0, x, 0);
}
